// MultiHeadAttention_63505386439259
// MI455X (gfx1250) — matrix operand order repaired and checked by reading, not compiled, not run on hardware
//
#include <hip/hip_runtime.h>
#include <hip/hip_bf16.h>
#include <math.h>

// ---------------------------------------------------------------------------
// MHA: B=2, S=2048, D=1024, H=16, dh=64. fp32 in/out, f16 WMMA compute.
// Round 6: modulo-scheduled GEMM loop (4 rotating buffers, unroll 4,
//          distance-2 prefetch) -> fixed per-position registers, no copies,
//          staggered load waits.
// ---------------------------------------------------------------------------

typedef _Float16 v16h __attribute__((ext_vector_type(16)));
typedef _Float16 v8h  __attribute__((ext_vector_type(8)));
typedef _Float16 v4h  __attribute__((ext_vector_type(4)));
typedef float    v8f  __attribute__((ext_vector_type(8)));

#define B_   2
#define S_   2048
#define D_   1024
#define H_   16
#define DH_  64

__device__ __forceinline__ v16h join8(v8h lo, v8h hi) {
    return __builtin_shufflevector(lo, hi, 0,1,2,3,4,5,6,7,8,9,10,11,12,13,14,15);
}

__device__ __forceinline__ v8f wmma16(v16h a, v16h b, v8f c) {
    // D = A(16x32 f16) * B(32x16 f16) + C(16x16 f32)
    return __builtin_amdgcn_wmma_f32_16x16x32_f16(false, a, false, b, (short)0, c,
                                                  false, false);
}

// Load one k-chunk of fragments: A (16x32) + 4x B (32x16).
__device__ __forceinline__ void load_frags(const _Float16* __restrict__ Arow,
                                           const _Float16* __restrict__ Brow,
                                           int k, v16h& a, v16h b[4]) {
    a = join8(*(const v8h*)(Arow + k), *(const v8h*)(Arow + k + 8));
#pragma unroll
    for (int t = 0; t < 4; ++t)
        b[t] = *(const v16h*)(Brow + (size_t)t * 16 * D_ + k);
}

// Modulo-scheduled 16x64 GEMM strip: 4 rotating buffers, unroll-by-4 loop,
// distance-2 prefetch. Buffer index is constant per unrolled position and
// repeats with the loop period -> no back-edge register copies.
__device__ __forceinline__ void gemm_strip(const _Float16* __restrict__ Arow,
                                           const _Float16* __restrict__ Brow,
                                           v8f acc[4]) {
    const int NC = D_ / 32;                    // 32 k-chunks
    v16h a[4];
    v16h b[4][4];
    load_frags(Arow, Brow, 0, a[0], b[0]);
    load_frags(Arow, Brow, 32, a[1], b[1]);
#pragma unroll 4
    for (int c = 0; c < NC - 4; ++c) {         // 28 iterations, 28 % 4 == 0
        load_frags(Arow, Brow, (c + 2) * 32, a[(c + 2) & 3], b[(c + 2) & 3]);
#pragma unroll
        for (int t = 0; t < 4; ++t)
            acc[t] = wmma16(a[c & 3], b[c & 3][t], acc[t]);
    }
    // epilogue: chunks NC-4 .. NC-1 (buffers 0..3 already/now filled)
    load_frags(Arow, Brow, (NC - 2) * 32, a[(NC - 2) & 3], b[(NC - 2) & 3]);
    load_frags(Arow, Brow, (NC - 1) * 32, a[(NC - 1) & 3], b[(NC - 1) & 3]);
#pragma unroll
    for (int c = NC - 4; c < NC; ++c) {
#pragma unroll
        for (int t = 0; t < 4; ++t)
            acc[t] = wmma16(a[c & 3], b[c & 3][t], acc[t]);
    }
}

// ---------------------------------------------------------------------------
// fp32 -> f16 conversion, 4 elems / thread
// ---------------------------------------------------------------------------
__global__ void cvt_f32_f16_x4(const float4* __restrict__ src,
                               _Float16* __restrict__ dst, int n4) {
    int i = blockIdx.x * blockDim.x + threadIdx.x;
    if (i < n4) {
        float4 v = src[i];
        v4h h = {(_Float16)v.x, (_Float16)v.y, (_Float16)v.z, (_Float16)v.w};
        *(v4h*)(dst + (size_t)i * 4) = h;
    }
}

// ---------------------------------------------------------------------------
// Fused QKV projection:  Y = x @ W^T  for W in {Wq, Wk, Wv}
//   Q,K written [B,H,S,dh];  V written transposed [B,H,dh,S].
// ---------------------------------------------------------------------------
__global__ __launch_bounds__(256)
__attribute__((amdgpu_waves_per_eu(2))) void qkv_gemm(
    const _Float16* __restrict__ xh, const _Float16* __restrict__ Wh,
    _Float16* __restrict__ Qh, _Float16* __restrict__ Kh,
    _Float16* __restrict__ Vt) {
    const int lane = threadIdx.x & 31;
    const int wave = threadIdx.x >> 5;
    const int l16  = lane & 15;
    const int hlf  = lane >> 4;          // half-wave id (0/1)

    int id  = blockIdx.x * 8 + wave;     // 0 .. 12287
    int w   = id >> 12;                  // weight select: 0=Q,1=K,2=V
    int r   = id & 4095;
    int m16 = r & 255;                   // row tile (16 rows)
    int n64 = r >> 8;                    // col tile (64 cols)

    const _Float16* Wp   = Wh + (size_t)w * (D_ * D_);
    const int       mrow = m16 * 16 + l16;
    const _Float16* Arow = xh + (size_t)mrow * D_ + hlf * 16;
    const _Float16* Brow = Wp + (size_t)(n64 * 64 + l16) * D_ + hlf * 16;

    v8f acc[4] = {};
    gemm_strip(Arow, Brow, acc);

    const int hb = hlf * 8;
#pragma unroll
    for (int t = 0; t < 4; ++t) {
        int n = n64 * 64 + t * 16 + l16;
        int h = n >> 6, d = n & 63;
#pragma unroll
        for (int i = 0; i < 8; ++i) {
            int m = m16 * 16 + hb + i;
            int b = m >> 11, s = m & (S_ - 1);
            _Float16 val = (_Float16)acc[t][i];
            if (w == 0)
                Qh[(((size_t)(b * H_ + h)) * S_ + s) * DH_ + d] = val;
            else if (w == 1)
                Kh[(((size_t)(b * H_ + h)) * S_ + s) * DH_ + d] = val;
            else
                Vt[(((size_t)(b * H_ + h)) * DH_ + d) * S_ + s] = val;
        }
    }
}

// ---------------------------------------------------------------------------
// Causal flash attention. 8 waves/block, 16 query rows per wave.
// All K/V fragment loads for a k-chunk are issued up front so the V loads'
// latency hides under the softmax VALU/shuffle work.
// ---------------------------------------------------------------------------
__global__ __launch_bounds__(256)
__attribute__((amdgpu_waves_per_eu(2))) void attn_kernel(
    const _Float16* __restrict__ Qh, const _Float16* __restrict__ Kh,
    const _Float16* __restrict__ Vt, _Float16* __restrict__ Ah) {
    __shared__ _Float16 pbuf[8][16 * 32];      // 1 KB per wave

    const int lane = threadIdx.x & 31;
    const int wave = threadIdx.x >> 5;
    const int l16  = lane & 15;
    const int hlf  = lane >> 4;
    const int hb   = hlf * 8;

    const int bh   = blockIdx.x >> 4;          // b*H + h  (0..31)
    const int qblk = blockIdx.x & 15;
    const int q0   = qblk * 128 + wave * 16;

    const _Float16* Qbase = Qh + (size_t)bh * S_ * DH_;
    const _Float16* Krow  = Kh + (size_t)bh * S_ * DH_ + (size_t)l16 * DH_ +
                            hlf * 16;          // lane's K column base
    const _Float16* Vrow  = Vt + (size_t)bh * DH_ * S_ + (size_t)l16 * S_ +
                            hlf * 16;          // lane's V column base (t=0)
    _Float16*       pw    = &pbuf[wave][0];

    // Preload Q A-fragments for the two 32-wide d-chunks.
    v16h aq[2];
    {
        const _Float16* qrow = Qbase + (size_t)(q0 + l16) * DH_ + hlf * 16;
#pragma unroll
        for (int c = 0; c < 2; ++c)
            aq[c] = join8(*(const v8h*)(qrow + c * 32),
                          *(const v8h*)(qrow + c * 32 + 8));
    }

    v8f   O[4] = {};
    float mrow[8], lrow[8];
#pragma unroll
    for (int i = 0; i < 8; ++i) { mrow[i] = -INFINITY; lrow[i] = 0.0f; }

    const int qmax = q0 + 15;
    for (int kb = 0; kb <= qmax; kb += 32) {
        // ---- issue ALL VMEM for this chunk up front ----
        v16h bk[2][2];                         // [sub][d-chunk]
#pragma unroll
        for (int sub = 0; sub < 2; ++sub) {
            const _Float16* kp = Krow + (size_t)(kb + sub * 16) * DH_;
            bk[sub][0] = *(const v16h*)(kp);
            bk[sub][1] = *(const v16h*)(kp + 32);
        }
        v16h bv[4];
#pragma unroll
        for (int t = 0; t < 4; ++t)
            bv[t] = *(const v16h*)(Vrow + (size_t)t * 16 * S_ + kb);

        // ---- scores for both 16-wide sub-tiles ----
        v8f sc[2];
#pragma unroll
        for (int sub = 0; sub < 2; ++sub) {
            v8f s0 = {};
            s0 = wmma16(aq[0], bk[sub][0], s0);
            s0 = wmma16(aq[1], bk[sub][1], s0);
            sc[sub] = s0;
        }

        // ---- online softmax (V loads still in flight) ----
#pragma unroll
        for (int sub = 0; sub < 2; ++sub) {
            const int col = kb + sub * 16 + l16;
#pragma unroll
            for (int i = 0; i < 8; ++i) {
                float s    = sc[sub][i] * 0.125f;        // 1/sqrt(64)
                int   qrow = q0 + hb + i;
                if (col > qrow) s = -INFINITY;           // causal mask
                float rm = s;
                rm = fmaxf(rm, __shfl_xor(rm, 1));
                rm = fmaxf(rm, __shfl_xor(rm, 2));
                rm = fmaxf(rm, __shfl_xor(rm, 4));
                rm = fmaxf(rm, __shfl_xor(rm, 8));
                float mnew  = fmaxf(mrow[i], rm);
                float mm    = (mnew == -INFINITY) ? 0.0f : mnew;
                float p     = __expf(s - mm);
                float scale = __expf(mrow[i] - mm);
                float ps = p;
                ps += __shfl_xor(ps, 1);
                ps += __shfl_xor(ps, 2);
                ps += __shfl_xor(ps, 4);
                ps += __shfl_xor(ps, 8);
                lrow[i] = lrow[i] * scale + ps;
                mrow[i] = mnew;
                O[0][i] *= scale; O[1][i] *= scale;
                O[2][i] *= scale; O[3][i] *= scale;
                pw[(hb + i) * 32 + sub * 16 + l16] = (_Float16)p;
            }
        }

        // ---- re-layout P(16x32) into A-fragment striping via LDS ----
        v16h pf = join8(*(const v8h*)(pw + l16 * 32 + hlf * 16),
                        *(const v8h*)(pw + l16 * 32 + 8 + hlf * 16));
#pragma unroll
        for (int t = 0; t < 4; ++t)
            O[t] = wmma16(pf, bv[t], O[t]);
    }

    const int b = bh >> 4, h = bh & 15;
#pragma unroll
    for (int i = 0; i < 8; ++i) {
        float inv = 1.0f / lrow[i];
        int   q   = q0 + hb + i;
#pragma unroll
        for (int t = 0; t < 4; ++t) {
            Ah[((size_t)(b * S_ + q)) * D_ + h * DH_ + t * 16 + l16] =
                (_Float16)(O[t][i] * inv);
        }
    }
}

// ---------------------------------------------------------------------------
// Output projection: out(f32) = Ah(f16) @ Wo^T(f16)
// ---------------------------------------------------------------------------
__global__ __launch_bounds__(256)
__attribute__((amdgpu_waves_per_eu(2))) void out_gemm(
    const _Float16* __restrict__ Ahh, const _Float16* __restrict__ Woh,
    float* __restrict__ out) {
    const int lane = threadIdx.x & 31;
    const int wave = threadIdx.x >> 5;
    const int l16  = lane & 15;
    const int hlf  = lane >> 4;

    int id  = blockIdx.x * 8 + wave;     // 0 .. 4095
    int m16 = id & 255;
    int n64 = id >> 8;

    const int       mrow = m16 * 16 + l16;
    const _Float16* Arow = Ahh + (size_t)mrow * D_ + hlf * 16;
    const _Float16* Brow = Woh + (size_t)(n64 * 64 + l16) * D_ + hlf * 16;

    v8f acc[4] = {};
    gemm_strip(Arow, Brow, acc);

    const int hb = hlf * 8;
#pragma unroll
    for (int t = 0; t < 4; ++t) {
        int n = n64 * 64 + t * 16 + l16;
#pragma unroll
        for (int i = 0; i < 8; ++i) {
            int m = m16 * 16 + hb + i;
            out[(size_t)m * D_ + n] = acc[t][i];
        }
    }
}

// ---------------------------------------------------------------------------
extern "C" void kernel_launch(void* const* d_in, const int* in_sizes, int n_in,
                              void* d_out, int out_size, void* d_ws,
                              size_t ws_size, hipStream_t stream) {
    (void)in_sizes; (void)n_in; (void)out_size; (void)ws_size;

    const float* x  = (const float*)d_in[0];
    const float* Wq = (const float*)d_in[1];
    const float* Wk = (const float*)d_in[2];
    const float* Wv = (const float*)d_in[3];
    const float* Wo = (const float*)d_in[4];

    const size_t NX = (size_t)B_ * S_ * D_;   // 4,194,304
    const size_t NW = (size_t)D_ * D_;        // 1,048,576

    _Float16* wsh = (_Float16*)d_ws;
    _Float16* xh  = wsh;                      //  8 MB
    _Float16* Wh  = wsh + NX;                 //  8 MB (4 weights back-to-back)
    _Float16* Qh  = wsh + 2 * NX;             //  8 MB  [B,H,S,dh]
    _Float16* Kh  = wsh + 3 * NX;             //  8 MB  [B,H,S,dh]
    _Float16* Vt  = wsh + 4 * NX;             //  8 MB  [B,H,dh,S]
    _Float16* Ah  = wsh + 5 * NX;             //  8 MB  [B,S,D]

    // 1) fp32 -> f16 conversions (4 elems / thread)
    cvt_f32_f16_x4<<<(int)(NX / 4 / 256), 256, 0, stream>>>(
        (const float4*)x, xh, (int)(NX / 4));
    cvt_f32_f16_x4<<<(int)(NW / 4 / 256), 256, 0, stream>>>(
        (const float4*)Wq, Wh + 0 * NW, (int)(NW / 4));
    cvt_f32_f16_x4<<<(int)(NW / 4 / 256), 256, 0, stream>>>(
        (const float4*)Wk, Wh + 1 * NW, (int)(NW / 4));
    cvt_f32_f16_x4<<<(int)(NW / 4 / 256), 256, 0, stream>>>(
        (const float4*)Wv, Wh + 2 * NW, (int)(NW / 4));
    cvt_f32_f16_x4<<<(int)(NW / 4 / 256), 256, 0, stream>>>(
        (const float4*)Wo, Wh + 3 * NW, (int)(NW / 4));

    // 2) fused Q/K/V projection (12288 wave-tiles / 8 waves per block)
    qkv_gemm<<<1536, 256, 0, stream>>>(xh, Wh, Qh, Kh, Vt);

    // 3) causal flash attention (B*H*(S/128) = 512 blocks)
    attn_kernel<<<512, 256, 0, stream>>>(Qh, Kh, Vt, Ah);

    // 4) output projection (4096 wave-tiles / 8 per block)
    out_gemm<<<512, 256, 0, stream>>>(Ah, Wh + 3 * NW, (float*)d_out);
}
